// VelPreVecPot3D_18021682774229
// MI455X (gfx1250) — matrix operand order repaired and checked by reading, not compiled, not run on hardware
//
#include <hip/hip_runtime.h>
#include <hip/hip_bf16.h>
#include <math.h>

// ---------------- types ----------------
typedef _Float16 v8h  __attribute__((ext_vector_type(8)));
typedef _Float16 v16h __attribute__((ext_vector_type(16)));
typedef float    v8f  __attribute__((ext_vector_type(8)));

#define DEPTH   512
#define NLAYERS 7          // WIDTH-1 hidden tanh blocks
#define MT      32         // samples per block
#define STR     520        // padded f16 row stride (1040B -> 4-bank row skew)
#define NSTREAM 4          // primal + 3 tangents
#define NT      4          // N-tiles accumulated per wave pass (64 columns)
#define BUF_ELEMS (MT * STR)

// ---------------- fast tanh ----------------
__device__ __forceinline__ float fast_tanh(float z) {
#if __has_builtin(__builtin_amdgcn_tanhf)
    return __builtin_amdgcn_tanhf(z);   // v_tanh_f32 (gfx1250 trans op)
#else
    return tanhf(z);
#endif
}

// ---------------- WMMA helper ----------------
__device__ __forceinline__ v8f wmma_f16(v16h a, v16h b, v8f c) {
    // D = A(16x32 f16) x B(32x16 f16) + C(16x16 f32)
    return __builtin_amdgcn_wmma_f32_16x16x32_f16(
        /*neg_a=*/false, a, /*neg_b=*/false, b,
        /*c_mod=*/(short)0, c, /*reuse_a=*/false, /*reuse_b=*/false);
}

// A fragment: rows m0..m0+15, K columns k0..k0+31 of LDS activation buffer.
// Layout (contiguous, matches B): lanes 0-15 hold K[0..15]; lanes 16-31
// hold K[16..31] for row M = lane%16.
__device__ __forceinline__ v16h load_a(const _Float16* base, int k0) {
    const _Float16* p = base + k0;
    v8h lo = *(const v8h*)p;                     // K = k0+kofs .. +7
    v8h hi = *(const v8h*)(p + 8);               // K = k0+kofs+8 .. +15
    return __builtin_shufflevector(lo, hi, 0,1,2,3,4,5,6,7,8,9,10,11,12,13,14,15);
}

// B fragment: W[k0..k0+31][ncol] from pre-transposed f16 weights Wt[n][k]
// (row-major, stride DEPTH). Layout used here (contiguous, matches A): lane L serves
// column N = L%16, K base (L/16)*16, VGPR v holds K = base+2v,2v+1
// -> 16 contiguous f16 = one 32B load.
__device__ __forceinline__ v16h load_b(const _Float16* colbase, int k0) {
    return *(const v16h*)(colbase + k0);
}

// ---------------- weight prep: Wh[L][k][n] (f32) -> Wt[L][n][k] (f16) -------
__global__ __launch_bounds__(256) void prep_weights_kernel(
    const float* __restrict__ Wh, _Float16* __restrict__ Wt) {
    int idx = blockIdx.x * 256 + threadIdx.x;    // one quad of k per thread
    const int total = NLAYERS * DEPTH * (DEPTH / 4);
    if (idx >= total) return;
    int k4 = (idx & 127) << 2;
    int n  = (idx >> 7) & 511;
    int L  = idx >> 16;                          // 512*128 = 65536 quads/layer
    const float* src = Wh + (size_t)L * DEPTH * DEPTH;
    _Float16*    dst = Wt + (size_t)L * DEPTH * DEPTH + (size_t)n * DEPTH + k4;
    dst[0] = (_Float16)src[(size_t)(k4 + 0) * DEPTH + n];
    dst[1] = (_Float16)src[(size_t)(k4 + 1) * DEPTH + n];
    dst[2] = (_Float16)src[(size_t)(k4 + 2) * DEPTH + n];
    dst[3] = (_Float16)src[(size_t)(k4 + 3) * DEPTH + n];
}

// ---------------- fused forward-mode MLP + curl -----------------------------
__global__ __launch_bounds__(256) void mlp_curl_kernel(
    const float* __restrict__ x,   const float* __restrict__ W0,
    const float* __restrict__ b0,  const _Float16* __restrict__ Wt,
    const float* __restrict__ bh,  const float* __restrict__ Wf,
    float* __restrict__ out) {
    extern __shared__ _Float16 smem[];
    _Float16* setA = smem;                         // 4 stream buffers
    _Float16* setB = smem + NSTREAM * BUF_ELEMS;   // ping-pong set

    const int tid  = threadIdx.x;
    const int lane = tid & 31;
    const int w    = tid >> 5;                     // wave id 0..7
    const int row0 = blockIdx.x * MT;

    // lane decodes (constant per thread)
    const int arow  = lane & 15;                   // A fragment row within tile
    const int akofs = (lane >> 4) << 4;            // A: K offset 0 or 16
    const int bcol  = lane & 15;                   // B fragment column
    const int bkofs = (lane >> 4) << 4;            // B: K offset 0 or 16
    const int ccol  = lane & 15;                   // C/D: N = lane%16
    const int cmb   = (lane >> 4) << 3;            // C/D: M base 0 or 8

    // ---- layer 0: h0 = x@W0 + b0 ; tangents g_k = W0 row k (no tanh) ----
    for (int idx = tid; idx < MT * DEPTH; idx += 256) {
        int m = idx >> 9, n = idx & 511;
        float4 xv = ((const float4*)x)[row0 + m];
        float w0n = W0[0 * DEPTH + n];
        float w1n = W0[1 * DEPTH + n];
        float w2n = W0[2 * DEPTH + n];
        float w3n = W0[3 * DEPTH + n];
        float h0 = b0[n] + xv.x * w0n + xv.y * w1n + xv.z * w2n + xv.w * w3n;
        setA[0 * BUF_ELEMS + m * STR + n] = (_Float16)h0;
        setA[1 * BUF_ELEMS + m * STR + n] = (_Float16)w0n;   // d h0 / d x0
        setA[2 * BUF_ELEMS + m * STR + n] = (_Float16)w1n;   // d h0 / d x1
        setA[3 * BUF_ELEMS + m * STR + n] = (_Float16)w2n;   // d h0 / d x2
    }
    __syncthreads();

    _Float16* cur = setA;
    _Float16* nxt = setB;

    // ---- 7 hidden layers: t=tanh(h@W+b); h'=t ; g'=(1-t^2) * (g@W) ----
    for (int L = 0; L < NLAYERS; ++L) {
        const _Float16* wl = Wt + (size_t)L * DEPTH * DEPTH;
        const float*    bl = bh + L * DEPTH;

        // prefetch next layer's weights into the cache hierarchy
        if (L + 1 < NLAYERS) {
            const _Float16* nw = Wt + (size_t)(L + 1) * DEPTH * DEPTH;
            #pragma unroll
            for (int q = 0; q < 16; ++q)
                __builtin_prefetch((const void*)(nw + tid * 1024 + q * 64), 0, 1);
        }

        // 16 items: (msub in 0..1) x (nquad in 0..7, 64 cols); wave w takes
        // (0,w) and (1,w) -> identical B columns both passes (L0-friendly).
        for (int p = 0; p < 2; ++p) {
            const int msub = p;
            const int m0   = msub * 16;
            const int n0   = w * 64;

            // per-stream A base pointers (lane-resolved)
            const _Float16* aBase[NSTREAM];
            #pragma unroll
            for (int s = 0; s < NSTREAM; ++s)
                aBase[s] = cur + s * BUF_ELEMS + (m0 + arow) * STR + akofs;
            // per-ntile B column base pointers (lane-resolved)
            const _Float16* bBase[NT];
            #pragma unroll
            for (int nt = 0; nt < NT; ++nt)
                bBase[nt] = wl + (size_t)(n0 + nt * 16 + bcol) * DEPTH + bkofs;

            const v8f vzero = {0.f, 0.f, 0.f, 0.f, 0.f, 0.f, 0.f, 0.f};
            v8f acc[NSTREAM][NT];
            #pragma unroll
            for (int s = 0; s < NSTREAM; ++s)
                #pragma unroll
                for (int nt = 0; nt < NT; ++nt) acc[s][nt] = vzero;

            #pragma unroll 2
            for (int k0 = 0; k0 < DEPTH; k0 += 32) {
                v16h a0 = load_a(aBase[0], k0);
                v16h a1 = load_a(aBase[1], k0);
                v16h a2 = load_a(aBase[2], k0);
                v16h a3 = load_a(aBase[3], k0);
                #pragma unroll
                for (int nt = 0; nt < NT; ++nt) {
                    v16h bf = load_b(bBase[nt], k0);
                    acc[0][nt] = wmma_f16(a0, bf, acc[0][nt]);
                    acc[1][nt] = wmma_f16(a1, bf, acc[1][nt]);
                    acc[2][nt] = wmma_f16(a2, bf, acc[2][nt]);
                    acc[3][nt] = wmma_f16(a3, bf, acc[3][nt]);
                }
            }

            // epilogue: C/D layout => element r of lane: M=cmb+r, N=ccol
            #pragma unroll
            for (int nt = 0; nt < NT; ++nt) {
                int n = n0 + nt * 16 + ccol;
                float bias = bl[n];
                #pragma unroll
                for (int r = 0; r < 8; ++r) {
                    int m = m0 + cmb + r;
                    float t  = fast_tanh(acc[0][nt][r] + bias);
                    float sc = 1.0f - t * t;
                    nxt[0 * BUF_ELEMS + m * STR + n] = (_Float16)t;
                    nxt[1 * BUF_ELEMS + m * STR + n] = (_Float16)(sc * acc[1][nt][r]);
                    nxt[2 * BUF_ELEMS + m * STR + n] = (_Float16)(sc * acc[2][nt][r]);
                    nxt[3 * BUF_ELEMS + m * STR + n] = (_Float16)(sc * acc[3][nt][r]);
                }
            }
        }
        __syncthreads();
        _Float16* t = cur; cur = nxt; nxt = t;
    }

    // ---- final: Jt[k][j] = sum_n g_k[n] * Wf[n][j], then curl assembly ----
    {
        int m    = tid >> 3;      // sample within tile
        int part = tid & 7;       // 64-wide n slice
        float J[12];
        #pragma unroll
        for (int j = 0; j < 12; ++j) J[j] = 0.f;
        int nbeg = part * 64;
        for (int n = nbeg; n < nbeg + 64; ++n) {
            float4 wf = ((const float4*)Wf)[n];
            float g0 = (float)cur[1 * BUF_ELEMS + m * STR + n];
            float g1 = (float)cur[2 * BUF_ELEMS + m * STR + n];
            float g2 = (float)cur[3 * BUF_ELEMS + m * STR + n];
            J[0] += g0 * wf.x; J[1] += g0 * wf.y; J[2]  += g0 * wf.z; J[3]  += g0 * wf.w;
            J[4] += g1 * wf.x; J[5] += g1 * wf.y; J[6]  += g1 * wf.z; J[7]  += g1 * wf.w;
            J[8] += g2 * wf.x; J[9] += g2 * wf.y; J[10] += g2 * wf.z; J[11] += g2 * wf.w;
        }
        float* scratch = (float*)nxt;   // dead ping-pong set reused as f32 scratch
        #pragma unroll
        for (int j = 0; j < 12; ++j) scratch[(m * 8 + part) * 12 + j] = J[j];
    }
    __syncthreads();
    if (tid < MT) {
        const float* scratch = (const float*)nxt;
        int m = tid;
        float J[12];
        #pragma unroll
        for (int j = 0; j < 12; ++j) J[j] = 0.f;
        for (int part = 0; part < 8; ++part)
            #pragma unroll
            for (int j = 0; j < 12; ++j) J[j] += scratch[(m * 8 + part) * 12 + j];
        // Jt[k][j] = J[k*4+j] = d a_j / d x_k
        float* o = out + (size_t)(row0 + m) * 6;
        o[0] = J[1 * 4 + 2] - J[2 * 4 + 1];   // da2/dx1 - da1/dx2
        o[1] = J[2 * 4 + 0] - J[0 * 4 + 2];   // da0/dx2 - da2/dx0
        o[2] = J[0 * 4 + 1] - J[1 * 4 + 0];   // da1/dx0 - da0/dx1
        o[3] = J[0 * 4 + 3];                  // dp/dx0
        o[4] = J[1 * 4 + 3];                  // dp/dx1
        o[5] = J[2 * 4 + 3];                  // dp/dx2
    }
}

// ---------------- host entry ----------------
extern "C" void kernel_launch(void* const* d_in, const int* in_sizes, int n_in,
                              void* d_out, int out_size, void* d_ws, size_t ws_size,
                              hipStream_t stream) {
    const float* x  = (const float*)d_in[0];   // [32768,4]
    const float* W0 = (const float*)d_in[1];   // [4,512]
    const float* b0 = (const float*)d_in[2];   // [512]
    const float* Wh = (const float*)d_in[3];   // [7,512,512]
    const float* bh = (const float*)d_in[4];   // [7,512]
    const float* Wf = (const float*)d_in[5];   // [512,4]
    float* out = (float*)d_out;                // [32768,6]
    _Float16* Wt = (_Float16*)d_ws;            // 7*512*512 f16 = 3.67 MB

    (void)in_sizes; (void)n_in; (void)out_size; (void)ws_size;

    // 1) transpose + downconvert hidden weights once
    const int quads = NLAYERS * DEPTH * (DEPTH / 4);
    prep_weights_kernel<<<(quads + 255) / 256, 256, 0, stream>>>(Wh, Wt);

    // 2) fused forward-mode MLP + curl (266,240 B dynamic LDS per workgroup)
    const size_t shmem = (size_t)2 * NSTREAM * BUF_ELEMS * sizeof(_Float16);
    hipFuncSetAttribute(reinterpret_cast<const void*>(mlp_curl_kernel),
                        hipFuncAttributeMaxDynamicSharedMemorySize, (int)shmem);
    mlp_curl_kernel<<<(32768 / MT), 256, shmem, stream>>>(x, W0, b0, Wt, bh, Wf, out);
}
